// LSTMLayerNorm_72980084293743
// MI455X (gfx1250) — matrix operand order repaired and checked by reading, not compiled, not run on hardware
//
#include <hip/hip_runtime.h>

#define NL 4
#define BB 16
#define TT 512
#define DD 1024
#define G4 4096
#define LN_EPS 1e-3f

typedef __bf16 bf16t;
typedef __attribute__((ext_vector_type(16))) __bf16 v16bf;
typedef __attribute__((ext_vector_type(8)))  float  v8f;

union FragBF { v16bf v; uint4 q[2]; };

__device__ __forceinline__ float sigf(float x) { return 1.0f / (1.0f + __expf(-x)); }

__device__ __forceinline__ void block_reduce2(float& a, float& b) {
  __shared__ float sa[256];
  __shared__ float sb[256];
  const int tid = threadIdx.x;
  sa[tid] = a; sb[tid] = b;
  __syncthreads();
  for (int o = 128; o > 0; o >>= 1) {
    if (tid < o) { sa[tid] += sa[tid + o]; sb[tid] += sb[tid + o]; }
    __syncthreads();
  }
  a = sa[0]; b = sb[0];
  __syncthreads();
}

// ---- precision / layout conversion kernels -------------------------------

// fp32 [DD, G4] -> bf16 transposed [G4, DD]
__global__ __launch_bounds__(256) void k_conv_wt(const float* __restrict__ W,
                                                 bf16t* __restrict__ Wt) {
  size_t i = (size_t)blockIdx.x * 256 + threadIdx.x;   // over G4*DD
  size_t n = i / DD, k = i - n * DD;
  Wt[i] = (bf16t)W[k * G4 + n];
}

// fp32 -> bf16 straight copy
__global__ __launch_bounds__(256) void k_conv_x(const float* __restrict__ x,
                                                bf16t* __restrict__ y) {
  size_t i = (size_t)blockIdx.x * 256 + threadIdx.x;
  y[i] = (bf16t)x[i];
}

// ---- big GEMM: C[M,N] = A[M,K] * Bt[N,K]^T  (bf16 in, f32 out) -----------
// grid = (N/512, M/16), block = 256 (8 waves); each wave owns 16x64 of C.

__global__ __launch_bounds__(256) void k_gemm_xw(
    const bf16t* __restrict__ A, const bf16t* __restrict__ Bt,
    float* __restrict__ C, int M, int N, int K) {
  const int lane = threadIdx.x & 31;
  const int wave = threadIdx.x >> 5;
  const int hi = lane >> 4;
  const int l15 = lane & 15;
  const int mrow = blockIdx.y << 4;
  const int ncol = (blockIdx.x * 8 + wave) << 6;

  const bf16t* pa  = A  + (size_t)(mrow + l15) * K + (hi ? 16 : 0);
  const bf16t* pb0 = Bt + (size_t)(ncol + l15) * K + (hi ? 16 : 0);
  const bf16t* pb1 = pb0 + (size_t)16 * K;
  const bf16t* pb2 = pb0 + (size_t)32 * K;
  const bf16t* pb3 = pb0 + (size_t)48 * K;

  v8f acc0 = {}, acc1 = {}, acc2 = {}, acc3 = {};
  for (int k0 = 0; k0 < K; k0 += 32) {
    FragBF a, b0, b1, b2, b3;
    a.q[0]  = *(const uint4*)(pa  + k0);
    a.q[1]  = *(const uint4*)(pa  + k0 + 8);
    b0.q[0] = *(const uint4*)(pb0 + k0);
    b0.q[1] = *(const uint4*)(pb0 + k0 + 8);
    b1.q[0] = *(const uint4*)(pb1 + k0);
    b1.q[1] = *(const uint4*)(pb1 + k0 + 8);
    b2.q[0] = *(const uint4*)(pb2 + k0);
    b2.q[1] = *(const uint4*)(pb2 + k0 + 8);
    b3.q[0] = *(const uint4*)(pb3 + k0);
    b3.q[1] = *(const uint4*)(pb3 + k0 + 8);
    __builtin_prefetch(pa  + k0 + 128, 0, 1);
    __builtin_prefetch(pb0 + k0 + 128, 0, 1);
    acc0 = __builtin_amdgcn_wmma_f32_16x16x32_bf16(false, a.v, false, b0.v, (short)0, acc0, false, false);
    acc1 = __builtin_amdgcn_wmma_f32_16x16x32_bf16(false, a.v, false, b1.v, (short)0, acc1, false, false);
    acc2 = __builtin_amdgcn_wmma_f32_16x16x32_bf16(false, a.v, false, b2.v, (short)0, acc2, false, false);
    acc3 = __builtin_amdgcn_wmma_f32_16x16x32_bf16(false, a.v, false, b3.v, (short)0, acc3, false, false);
  }
  const int rb = hi << 3;
  float* c0 = C + (size_t)(mrow + rb) * N + (ncol + l15);
#pragma unroll
  for (int r = 0; r < 8; ++r) {
    c0[(size_t)r * N +  0] = acc0[r];
    c0[(size_t)r * N + 16] = acc1[r];
    c0[(size_t)r * N + 32] = acc2[r];
    c0[(size_t)r * N + 48] = acc3[r];
  }
}

// ---- recurrent step GEMM: hU[16,4096] = h[16,1024] * Ut[4096,1024]^T -----
// grid = 32, block = 256; each wave owns one 16x16 tile (low latency).

__global__ __launch_bounds__(256) void k_step_gemm(
    const bf16t* __restrict__ A, const bf16t* __restrict__ Bt,
    float* __restrict__ C) {
  const int lane = threadIdx.x & 31;
  const int wave = threadIdx.x >> 5;
  const int hi = lane >> 4;
  const int l15 = lane & 15;
  const int ncol = (blockIdx.x * 8 + wave) << 4;
  const bf16t* pa = A  + (size_t)l15 * DD + (hi ? 16 : 0);
  const bf16t* pb = Bt + (size_t)(ncol + l15) * DD + (hi ? 16 : 0);
  v8f acc = {};
  for (int k0 = 0; k0 < DD; k0 += 32) {
    FragBF a, b;
    a.q[0] = *(const uint4*)(pa + k0);
    a.q[1] = *(const uint4*)(pa + k0 + 8);
    b.q[0] = *(const uint4*)(pb + k0);
    b.q[1] = *(const uint4*)(pb + k0 + 8);
    __builtin_prefetch(pb + k0 + 128, 0, 1);
    acc = __builtin_amdgcn_wmma_f32_16x16x32_bf16(false, a.v, false, b.v, (short)0, acc, false, false);
  }
  const int rb = hi << 3;
  float* c0 = C + (size_t)rb * G4 + (ncol + l15);
#pragma unroll
  for (int r = 0; r < 8; ++r) c0[(size_t)r * G4] = acc[r];
}

// ---- per-row LN stats for XW (mean, rstd) --------------------------------
__global__ __launch_bounds__(256) void k_row_stats(const float* __restrict__ X,
                                                   float* __restrict__ st) {
  const int row = blockIdx.x;
  const float* p = X + (size_t)row * G4;
  float s = 0.f, ss = 0.f;
  for (int j = threadIdx.x; j < G4; j += 256) { float v = p[j]; s += v; ss = fmaf(v, v, ss); }
  block_reduce2(s, ss);
  if (threadIdx.x == 0) {
    float m = s * (1.0f / G4);
    st[(size_t)row * 2]     = m;
    st[(size_t)row * 2 + 1] = rsqrtf(fmaxf(ss * (1.0f / G4) - m * m, 0.f) + LN_EPS);
  }
}

__global__ __launch_bounds__(256) void k_init_state(const float* __restrict__ st,
                                                    bf16t* __restrict__ hbf,
                                                    float* __restrict__ hf,
                                                    float* __restrict__ c) {
  int i = blockIdx.x * 256 + threadIdx.x;      // < BB*DD
  float h = st[i];
  hbf[i] = (bf16t)h; hf[i] = h; c[i] = st[BB * DD + i];
}

__device__ __forceinline__ float gatev(const float* xw, const float* hu,
    float mx, float rx, float mr, float rr,
    const float* gk, const float* bk, const float* gr, const float* br,
    const float* bl, int p) {
  return (xw[p] - mx) * rx * gk[p] + bk[p] + (hu[p] - mr) * rr * gr[p] + br[p] + bl[p];
}

// ---- fused: LN(hU) + gates + cell LN + h update; one WG per batch row ----
__global__ __launch_bounds__(256) void k_step_update(
    const float* __restrict__ hU, const float* __restrict__ XW,
    const float* __restrict__ xwst,
    const float* __restrict__ gk, const float* __restrict__ bk,
    const float* __restrict__ gr, const float* __restrict__ br,
    const float* __restrict__ bl,
    const float* __restrict__ gs, const float* __restrict__ bs,
    float* __restrict__ c, bf16t* __restrict__ hbf, float* __restrict__ hf,
    bf16t* __restrict__ yb, float* __restrict__ yf, int t) {
  const int b = blockIdx.x, tid = threadIdx.x;
  const float* hu = hU + (size_t)b * G4;

  float s = 0.f, ss = 0.f;
  for (int j = tid; j < G4; j += 256) { float v = hu[j]; s += v; ss = fmaf(v, v, ss); }
  block_reduce2(s, ss);
  const float mr = s * (1.0f / G4);
  const float rr = rsqrtf(fmaxf(ss * (1.0f / G4) - mr * mr, 0.f) + LN_EPS);

  const size_t row = (size_t)b * TT + t;
  const float* xw = XW + row * G4;
  const float mx = xwst[row * 2], rx = xwst[row * 2 + 1];

  float cn[4], zo[4];
#pragma unroll
  for (int u = 0; u < 4; ++u) {
    const int j = tid + (u << 8);
    float zi = gatev(xw, hu, mx, rx, mr, rr, gk, bk, gr, br, bl, j);
    float zf = gatev(xw, hu, mx, rx, mr, rr, gk, bk, gr, br, bl, j + 1024);
    float zg = gatev(xw, hu, mx, rx, mr, rr, gk, bk, gr, br, bl, j + 2048);
    zo[u]    = gatev(xw, hu, mx, rx, mr, rr, gk, bk, gr, br, bl, j + 3072);
    cn[u] = sigf(zf) * c[(size_t)b * DD + j] + sigf(zi) * tanhf(zg);
  }

  s = 0.f; ss = 0.f;
#pragma unroll
  for (int u = 0; u < 4; ++u) { s += cn[u]; ss = fmaf(cn[u], cn[u], ss); }
  block_reduce2(s, ss);
  const float ms = s * (1.0f / DD);
  const float rs = rsqrtf(fmaxf(ss * (1.0f / DD) - ms * ms, 0.f) + LN_EPS);

#pragma unroll
  for (int u = 0; u < 4; ++u) {
    const int j = tid + (u << 8);
    float cno = (cn[u] - ms) * rs * gs[j] + bs[j];
    c[(size_t)b * DD + j] = cno;
    float hn = sigf(zo[u]) * tanhf(cno);
    hf[(size_t)b * DD + j]  = hn;
    hbf[(size_t)b * DD + j] = (bf16t)hn;
    if (yf) yf[row * DD + j] = hn;
    else    yb[row * DD + j] = (bf16t)hn;
  }
}

__global__ __launch_bounds__(256) void k_finalize(const float* __restrict__ hf,
                                                  const float* __restrict__ c,
                                                  float* __restrict__ o) {
  int i = blockIdx.x * 256 + threadIdx.x;      // < BB*DD
  o[i] = hf[i]; o[BB * DD + i] = c[i];
}

// --------------------------------------------------------------------------

extern "C" void kernel_launch(void* const* d_in, const int* in_sizes, int n_in,
                              void* d_out, int out_size, void* d_ws, size_t ws_size,
                              hipStream_t stream) {
  (void)in_sizes; (void)n_in; (void)out_size; (void)ws_size;
  const float* inp  = (const float*)d_in[0];
  const float* st   = (const float*)d_in[1];
  const float* W    = (const float*)d_in[2];
  const float* U    = (const float*)d_in[3];
  const float* bias = (const float*)d_in[4];
  const float* g_k  = (const float*)d_in[5];
  const float* b_k  = (const float*)d_in[6];
  const float* g_r  = (const float*)d_in[7];
  const float* b_r  = (const float*)d_in[8];
  const float* g_s  = (const float*)d_in[9];
  const float* b_s  = (const float*)d_in[10];

  float* outX = (float*)d_out;
  float* outS = (float*)d_out + (size_t)BB * TT * DD;

  char* p = (char*)d_ws;
  auto take = [&](size_t bytes) { char* r = p; p += (bytes + 255) & ~(size_t)255; return r; };
  bf16t* Wt   = (bf16t*)take(sizeof(bf16t) * (size_t)G4 * DD);
  bf16t* Ut   = (bf16t*)take(sizeof(bf16t) * (size_t)G4 * DD);
  bf16t* Xp   = (bf16t*)take(sizeof(bf16t) * (size_t)BB * TT * DD);
  bf16t* Xq   = (bf16t*)take(sizeof(bf16t) * (size_t)BB * TT * DD);
  float* XW   = (float*)take(sizeof(float) * (size_t)BB * TT * G4);
  float* xwst = (float*)take(sizeof(float) * (size_t)BB * TT * 2);
  float* hU   = (float*)take(sizeof(float) * (size_t)BB * G4);
  bf16t* hbf  = (bf16t*)take(sizeof(bf16t) * (size_t)BB * DD);
  float* hf   = (float*)take(sizeof(float) * (size_t)BB * DD);
  float* cbuf = (float*)take(sizeof(float) * (size_t)BB * DD);

  const int M = BB * TT;                       // 8192 rows
  k_conv_x<<<(M * DD) / 256, 256, 0, stream>>>(inp, Xp);

  bf16t* cur = Xp; bf16t* nxt = Xq;
  for (int l = 0; l < NL; ++l) {
    k_conv_wt<<<(G4 * DD) / 256, 256, 0, stream>>>(W + (size_t)l * DD * G4, Wt);
    k_conv_wt<<<(G4 * DD) / 256, 256, 0, stream>>>(U + (size_t)l * DD * G4, Ut);
    k_gemm_xw<<<dim3(G4 / 512, M / 16), 256, 0, stream>>>(cur, Wt, XW, M, G4, DD);
    k_row_stats<<<M, 256, 0, stream>>>(XW, xwst);
    k_init_state<<<(BB * DD) / 256, 256, 0, stream>>>(st + (size_t)l * 2 * BB * DD, hbf, hf, cbuf);

    const float* gk = g_k + (size_t)l * G4;
    const float* bk = b_k + (size_t)l * G4;
    const float* gr = g_r + (size_t)l * G4;
    const float* br = b_r + (size_t)l * G4;
    const float* bl = bias + (size_t)l * G4;
    const float* gs = g_s + (size_t)l * DD;
    const float* bs = b_s + (size_t)l * DD;

    for (int t = 0; t < TT; ++t) {
      k_step_gemm<<<32, 256, 0, stream>>>(hbf, Ut, hU);
      k_step_update<<<BB, 256, 0, stream>>>(hU, XW, xwst, gk, bk, gr, br, bl, gs, bs,
                                            cbuf, hbf, hf,
                                            (l < NL - 1) ? nxt : (bf16t*)nullptr,
                                            (l == NL - 1) ? outX : (float*)nullptr, t);
    }
    k_finalize<<<(BB * DD) / 256, 256, 0, stream>>>(hf, cbuf, outS + (size_t)l * 2 * BB * DD);
    bf16t* tmp = cur; cur = nxt; nxt = tmp;
  }
}
